// GATv2Encoder_50259707298359
// MI455X (gfx1250) — compile-verified, matrix operand order checked by reading, not run on hardware
//
#include <hip/hip_runtime.h>
#include <math.h>

// ---------------- problem constants (match reference) ----------------
#define GAT_N      50000
#define GAT_E      800000
#define GAT_IN     128
#define GAT_HID    32
#define GAT_HEADS  4
#define GAT_HC     (GAT_HEADS * GAT_HID)   // 128
#define GAT_OUT    64
#define GAT_ED     32
#define NEG_SLOPE  0.2f

typedef __attribute__((ext_vector_type(16))) _Float16 v16h;
typedef __attribute__((ext_vector_type(8)))  _Float16 v8h;
typedef __attribute__((ext_vector_type(8)))  float    v8f;

// ---------------------------------------------------------------------
// WMMA tile loader: 16x32 f16 tile in the CDNA5 A-matrix VGPR layout.
// Row-major source with row stride `ld` (elements). Lane L<16 holds row
// (row_base+L), K = k0+0..7 and k0+16..23; lane L>=16 holds K = k0+8..15
// and k0+24..31.  Both halves are contiguous 16B loads (ld, k0 multiples
// of 8 -> 16B aligned given a 16B-aligned base).
// B (32x16, K x Ncol) is fed the same way from the *transposed* weight
// matrix Wt[Ncol x K], so A and B share one loader.
// ---------------------------------------------------------------------
__device__ __forceinline__ v16h load_tile_16x32(const _Float16* __restrict__ base,
                                                int ld, int row_base, int k0, int lane) {
  const int r  = lane & 15;
  const int ks = (lane >> 4) << 3;              // 0 or 8
  const _Float16* p = base + (size_t)(row_base + r) * ld + (k0 + ks);
  v8h lo = *(const v8h*)(p);                    // K = k0+ks   .. +7
  v8h hi = *(const v8h*)(p + 16);               // K = k0+16+ks.. +7
  v16h out;
#pragma unroll
  for (int i = 0; i < 8; ++i) { out[i] = lo[i]; out[i + 8] = hi[i]; }
  return out;
}

// ---------------------------------------------------------------------
// Dual GEMM: C0 = A@W0^T + b0,  C1 = A@W1^T + b1  (A: [M x K] f16,
// Wt*: [Ncol x K] f16, C*: [M x Ncol] f32).  One wave per 16x16 tile of
// both outputs; the A tile is loaded once and feeds two v_wmma chains.
// grid = (M/16, Ncol/16), block = 32 (one wave, EXEC all-ones for WMMA).
// ---------------------------------------------------------------------
__global__ void __launch_bounds__(32)
gemm_dual_wmma_kernel(const _Float16* __restrict__ A, int K,
                      const _Float16* __restrict__ Wt0, const float* __restrict__ b0,
                      float* __restrict__ C0,
                      const _Float16* __restrict__ Wt1, const float* __restrict__ b1,
                      float* __restrict__ C1,
                      int Ncol) {
  const int mb   = blockIdx.x * 16;
  const int nb   = blockIdx.y * 16;
  const int lane = threadIdx.x;

  v8f acc0 = {};
  v8f acc1 = {};
  for (int k0 = 0; k0 < K; k0 += 32) {
    v16h a  = load_tile_16x32(A,   K, mb, k0, lane);
    v16h w0 = load_tile_16x32(Wt0, K, nb, k0, lane);
    v16h w1 = load_tile_16x32(Wt1, K, nb, k0, lane);
    acc0 = __builtin_amdgcn_wmma_f32_16x16x32_f16(false, a, false, w0,
                                                  (short)0, acc0, false, false);
    acc1 = __builtin_amdgcn_wmma_f32_16x16x32_f16(false, a, false, w1,
                                                  (short)0, acc1, false, false);
  }

  // C/D layout: VGPR v -> row = mb + v + 8*(lane>=16), col = nb + (lane&15)
  const int col   = nb + (lane & 15);
  const int rbase = mb + ((lane >> 4) << 3);
  const float bv0 = b0[col];
  const float bv1 = b1[col];
#pragma unroll
  for (int v = 0; v < 8; ++v) {
    C0[(size_t)(rbase + v) * Ncol + col] = acc0[v] + bv0;
    C1[(size_t)(rbase + v) * Ncol + col] = acc1[v] + bv1;
  }
}

// ---------------------------- small utils ----------------------------
__global__ void cvt_f32_to_f16_kernel(const float* __restrict__ in,
                                      _Float16* __restrict__ out, size_t n) {
  size_t t = (size_t)blockIdx.x * blockDim.x + threadIdx.x;
  if (t < n) out[t] = (_Float16)in[t];
}

// Wt[n][k] = W[k][n]  (W: [K x Ncol] f32 -> Wt: [Ncol x K] f16)
__global__ void transpose_to_f16_kernel(const float* __restrict__ W,
                                        _Float16* __restrict__ Wt,
                                        int K, int Ncol) {
  int t = blockIdx.x * blockDim.x + threadIdx.x;
  if (t >= K * Ncol) return;
  int k = t / Ncol, n = t % Ncol;
  Wt[(size_t)n * K + k] = (_Float16)W[(size_t)k * Ncol + n];
}

__global__ void fill_f32_kernel(float* __restrict__ p, float v, size_t n) {
  size_t t = (size_t)blockIdx.x * blockDim.x + threadIdx.x;
  if (t < n) p[t] = v;
}

__device__ __forceinline__ void atomicMaxF(float* addr, float val) {
  unsigned int* u = (unsigned int*)addr;
  unsigned int old = __builtin_nontemporal_load(u);
  while (__uint_as_float(old) < val) {
    unsigned int assumed = old;
    old = atomicCAS(u, assumed, __float_as_uint(val));
    if (old == assumed) break;
  }
}

// ---------------------------------------------------------------------
// Edge phase 1: attention logits with on-the-fly edge transform.
//   a[e,h] = sum_c att[h,c] * leaky_relu(xl[src,h,c] + xr[dst,h,c]
//                                        + sum_k ea[e,k]*We[k,h*C+c])
// plus fused segment-max into amax[dst,h].  We (<=16KB) and att live in LDS.
// One thread per (edge, head).
// ---------------------------------------------------------------------
__global__ void edge_score_kernel(const int* __restrict__ src, const int* __restrict__ dst,
                                  const float* __restrict__ ea,
                                  const float* __restrict__ We, const float* __restrict__ att,
                                  const float* __restrict__ xl, const float* __restrict__ xr,
                                  float* __restrict__ alog, float* __restrict__ amax,
                                  int E_, int H, int C) {
  const int HC = H * C;
  __shared__ float sWe[GAT_ED * GAT_HC];   // 16 KB max
  __shared__ float sAtt[GAT_HC];
  for (int i = threadIdx.x; i < GAT_ED * HC; i += blockDim.x) sWe[i] = We[i];
  for (int i = threadIdx.x; i < HC; i += blockDim.x) sAtt[i] = att[i];
  __syncthreads();

  const int t = blockIdx.x * blockDim.x + threadIdx.x;
  if (t >= E_ * H) return;
  const int e = t / H, h = t % H;
  const int s = src[e], d = dst[e];

  float eav[GAT_ED];
#pragma unroll
  for (int k = 0; k < GAT_ED; ++k) eav[k] = ea[(size_t)e * GAT_ED + k];

  const float* pl = xl + (size_t)s * HC + h * C;
  const float* pr = xr + (size_t)d * HC + h * C;
  const float* pw = sWe + h * C;
  const float* pa = sAtt + h * C;

  float acc = 0.f;
  for (int c = 0; c < C; ++c) {
    float ee = 0.f;
#pragma unroll
    for (int k = 0; k < GAT_ED; ++k) ee = fmaf(eav[k], pw[k * HC + c], ee);
    float v = pl[c] + pr[c] + ee;
    v = (v > 0.f) ? v : NEG_SLOPE * v;
    acc = fmaf(pa[c], v, acc);
  }
  alog[t] = acc;
  atomicMaxF(&amax[(size_t)d * H + h], acc);
}

// Edge phase 2: ex = exp(a - amax[dst]); den[dst] += ex  (ex stored over a)
__global__ void edge_expsum_kernel(const int* __restrict__ dst,
                                   float* __restrict__ alog, const float* __restrict__ amax,
                                   float* __restrict__ den, int EH, int H) {
  const int t = blockIdx.x * blockDim.x + threadIdx.x;
  if (t >= EH) return;
  const int e = t / H, h = t % H;
  const int d = dst[e];
  const float ex = expf(alog[t] - amax[(size_t)d * H + h]);
  alog[t] = ex;
  atomicAdd(&den[(size_t)d * H + h], ex);
}

// Edge phase 3: out[dst,h,:] += xl[src,h,:] * (ex / (den[dst,h] + 1e-16))
__global__ void edge_scatter_kernel(const int* __restrict__ src, const int* __restrict__ dst,
                                    const float* __restrict__ xl,
                                    const float* __restrict__ ex, const float* __restrict__ den,
                                    float* __restrict__ out, int EH, int H, int C) {
  const int t = blockIdx.x * blockDim.x + threadIdx.x;
  if (t >= EH) return;
  const int e = t / H, h = t % H;
  const int s = src[e], d = dst[e];
  const float alpha = ex[t] / (den[(size_t)d * H + h] + 1e-16f);
  const float* pl = xl + (size_t)s * H * C + h * C;
  float* po = out + (size_t)d * H * C + h * C;
  for (int c = 0; c < C; ++c) atomicAdd(&po[c], pl[c] * alpha);
}

// Layer-1 epilogue: h = elu(acc + bias), kept fp32 (in place) + f16 copy
__global__ void finalize_elu_kernel(float* __restrict__ acc, const float* __restrict__ bias,
                                    _Float16* __restrict__ out16, size_t n, int HC) {
  size_t t = (size_t)blockIdx.x * blockDim.x + threadIdx.x;
  if (t >= n) return;
  float v = acc[t] + bias[t % HC];
  v = (v > 0.f) ? v : (expf(v) - 1.f);
  acc[t] = v;
  out16[t] = (_Float16)v;
}

// Layer-2 epilogue: out = acc + bias
__global__ void add_bias_kernel(const float* __restrict__ acc, const float* __restrict__ bias,
                                float* __restrict__ out, size_t n, int C) {
  size_t t = (size_t)blockIdx.x * blockDim.x + threadIdx.x;
  if (t >= n) return;
  out[t] = acc[t] + bias[t % C];
}

// ---------------------------------------------------------------------
static inline size_t align256(size_t x) { return (x + 255) & ~(size_t)255; }

extern "C" void kernel_launch(void* const* d_in, const int* in_sizes, int n_in,
                              void* d_out, int out_size, void* d_ws, size_t ws_size,
                              hipStream_t stream) {
  const float* x     = (const float*)d_in[0];
  const int*   eidx  = (const int*)  d_in[1];
  const float* ea    = (const float*)d_in[2];
  const float* Wl1   = (const float*)d_in[3];
  const float* bl1   = (const float*)d_in[4];
  const float* Wr1   = (const float*)d_in[5];
  const float* br1   = (const float*)d_in[6];
  const float* We1   = (const float*)d_in[7];
  const float* att1  = (const float*)d_in[8];
  const float* bias1 = (const float*)d_in[9];
  const float* Wl2   = (const float*)d_in[10];
  const float* bl2   = (const float*)d_in[11];
  const float* Wr2   = (const float*)d_in[12];
  const float* br2   = (const float*)d_in[13];
  const float* We2   = (const float*)d_in[14];
  const float* att2  = (const float*)d_in[15];
  const float* bias2 = (const float*)d_in[16];

  const int N = in_sizes[0] / GAT_IN;     // 50000
  const int E = in_sizes[1] / 2;          // 800000
  const int* src = eidx;
  const int* dst = eidx + E;

  // ---- workspace carve-up (~159 MB) ----
  char* ws = (char*)d_ws;
  size_t off = 0;
  auto carve = [&](size_t bytes) { char* p = ws + off; off = align256(off + bytes); return p; };

  _Float16* x16    = (_Float16*)carve((size_t)N * GAT_IN * 2);
  _Float16* h16    = (_Float16*)carve((size_t)N * GAT_HC * 2);
  _Float16* Wl1t   = (_Float16*)carve((size_t)GAT_HC * GAT_IN * 2);
  _Float16* Wr1t   = (_Float16*)carve((size_t)GAT_HC * GAT_IN * 2);
  _Float16* Wl2t   = (_Float16*)carve((size_t)GAT_OUT * GAT_HC * 2);
  _Float16* Wr2t   = (_Float16*)carve((size_t)GAT_OUT * GAT_HC * 2);
  float*    xl1    = (float*)carve((size_t)N * GAT_HC * 4);
  float*    xr1    = (float*)carve((size_t)N * GAT_HC * 4);
  float*    xl2    = (float*)carve((size_t)N * GAT_OUT * 4);
  float*    xr2    = (float*)carve((size_t)N * GAT_OUT * 4);
  float*    a1     = (float*)carve((size_t)E * GAT_HEADS * 4);
  float*    a2     = (float*)carve((size_t)E * 4);
  float*    amax1  = (float*)carve((size_t)N * GAT_HEADS * 4);
  float*    den1   = (float*)carve((size_t)N * GAT_HEADS * 4);
  float*    amax2  = (float*)carve((size_t)N * 4);
  float*    den2   = (float*)carve((size_t)N * 4);
  float*    hacc   = (float*)carve((size_t)N * GAT_HC * 4);   // layer1 scatter acc -> h
  float*    acc2   = (float*)carve((size_t)N * GAT_OUT * 4);  // layer2 scatter acc
  (void)ws_size; (void)n_in; (void)out_size;

  const int TB = 256;
  auto blocks = [&](size_t n) { return (unsigned)((n + TB - 1) / TB); };

  // ---- precision staging: x -> f16, weights -> transposed f16 ----
  cvt_f32_to_f16_kernel<<<blocks((size_t)N * GAT_IN), TB, 0, stream>>>(x, x16, (size_t)N * GAT_IN);
  transpose_to_f16_kernel<<<blocks(GAT_IN * GAT_HC), TB, 0, stream>>>(Wl1, Wl1t, GAT_IN, GAT_HC);
  transpose_to_f16_kernel<<<blocks(GAT_IN * GAT_HC), TB, 0, stream>>>(Wr1, Wr1t, GAT_IN, GAT_HC);
  transpose_to_f16_kernel<<<blocks(GAT_HC * GAT_OUT), TB, 0, stream>>>(Wl2, Wl2t, GAT_HC, GAT_OUT);
  transpose_to_f16_kernel<<<blocks(GAT_HC * GAT_OUT), TB, 0, stream>>>(Wr2, Wr2t, GAT_HC, GAT_OUT);

  // ---- layer 1 ----
  {
    dim3 g(N / 16, GAT_HC / 16);
    gemm_dual_wmma_kernel<<<g, 32, 0, stream>>>(x16, GAT_IN, Wl1t, bl1, xl1,
                                                Wr1t, br1, xr1, GAT_HC);
  }
  fill_f32_kernel<<<blocks((size_t)N * GAT_HEADS), TB, 0, stream>>>(amax1, -INFINITY, (size_t)N * GAT_HEADS);
  fill_f32_kernel<<<blocks((size_t)N * GAT_HEADS), TB, 0, stream>>>(den1, 0.f, (size_t)N * GAT_HEADS);
  fill_f32_kernel<<<blocks((size_t)N * GAT_HC), TB, 0, stream>>>(hacc, 0.f, (size_t)N * GAT_HC);

  edge_score_kernel<<<blocks((size_t)E * GAT_HEADS), TB, 0, stream>>>(
      src, dst, ea, We1, att1, xl1, xr1, a1, amax1, E, GAT_HEADS, GAT_HID);
  edge_expsum_kernel<<<blocks((size_t)E * GAT_HEADS), TB, 0, stream>>>(
      dst, a1, amax1, den1, E * GAT_HEADS, GAT_HEADS);
  edge_scatter_kernel<<<blocks((size_t)E * GAT_HEADS), TB, 0, stream>>>(
      src, dst, xl1, a1, den1, hacc, E * GAT_HEADS, GAT_HEADS, GAT_HID);
  finalize_elu_kernel<<<blocks((size_t)N * GAT_HC), TB, 0, stream>>>(
      hacc, bias1, h16, (size_t)N * GAT_HC, GAT_HC);

  // ---- layer 2 (H=1, C=64, concat=False -> mean over 1 head = identity) ----
  {
    dim3 g(N / 16, GAT_OUT / 16);
    gemm_dual_wmma_kernel<<<g, 32, 0, stream>>>(h16, GAT_HC, Wl2t, bl2, xl2,
                                                Wr2t, br2, xr2, GAT_OUT);
  }
  fill_f32_kernel<<<blocks((size_t)N), TB, 0, stream>>>(amax2, -INFINITY, (size_t)N);
  fill_f32_kernel<<<blocks((size_t)N), TB, 0, stream>>>(den2, 0.f, (size_t)N);
  fill_f32_kernel<<<blocks((size_t)N * GAT_OUT), TB, 0, stream>>>(acc2, 0.f, (size_t)N * GAT_OUT);

  edge_score_kernel<<<blocks((size_t)E), TB, 0, stream>>>(
      src, dst, ea, We2, att2, xl2, xr2, a2, amax2, E, 1, GAT_OUT);
  edge_expsum_kernel<<<blocks((size_t)E), TB, 0, stream>>>(dst, a2, amax2, den2, E, 1);
  edge_scatter_kernel<<<blocks((size_t)E), TB, 0, stream>>>(
      src, dst, xl2, a2, den2, acc2, E, 1, GAT_OUT);
  add_bias_kernel<<<blocks((size_t)N * GAT_OUT), TB, 0, stream>>>(
      acc2, bias2, (float*)d_out, (size_t)N * GAT_OUT, GAT_OUT);
}
